// RelationSimplestBlock_15788299780643
// MI455X (gfx1250) — matrix operand order repaired and checked by reading, not compiled, not run on hardware
//
#include <hip/hip_runtime.h>
#include <math.h>

// ---------------------------------------------------------------------------
// RelationSimplestBlock forward for MI455X (gfx1250), wave32 + WMMA f16.
// B=8, N=1024, D=256, H=8, DV=32, KD=256, HID=1024.
// Forward value of the straight-through estimator is exactly `hard`
// (soft - (soft - hard) == hard), and argmax(softmax(s)) == argmax(s),
// so we only need Q.K^T scores -> argmax -> gather V.
// ---------------------------------------------------------------------------

typedef __attribute__((ext_vector_type(16))) _Float16 v16h;
typedef __attribute__((ext_vector_type(8)))  _Float16 v8h;
typedef __attribute__((ext_vector_type(8)))  float    v8f;

#define RB   8        // batch
#define RN   1024     // seq
#define RD   256      // dim
#define RH   8        // heads
#define RDV  32       // head dim
#define RHID 1024     // RN hidden
#define ROWS (RB * RN)   // 8192

// ---------------------------------------------------------------------------
// Kernel 1: weight prep. Convert all weights to f16 stored TRANSPOSED [N,K]
// so WMMA B-tiles are contiguous 32B per lane (K fast-moving).
// ---------------------------------------------------------------------------
__global__ void rsb_prep_weights(const float* __restrict__ wq,
                                 const float* __restrict__ wkv,
                                 const float* __restrict__ wdown,
                                 const float* __restrict__ winner,
                                 const float* __restrict__ wouter,
                                 _Float16* __restrict__ wq_t,     // [32,256]
                                 _Float16* __restrict__ wkv_t,    // [512,256]
                                 _Float16* __restrict__ wdown_t,  // [32,256]
                                 _Float16* __restrict__ winner_t, // [1024,64]
                                 _Float16* __restrict__ wouter_t) // [256,1024]
{
    int i = blockIdx.x * blockDim.x + threadIdx.x;
    const int s0 = 32 * 256, s1 = 512 * 256, s2 = 32 * 256, s3 = 1024 * 64, s4 = 256 * 1024;
    if (i < s0) { int n = i / 256, k = i % 256; wq_t[i]     = (_Float16)wq[k * 32 + n];      return; }
    i -= s0;
    if (i < s1) { int n = i / 256, k = i % 256; wkv_t[i]    = (_Float16)wkv[k * 512 + n];    return; }
    i -= s1;
    if (i < s2) { int n = i / 256, k = i % 256; wdown_t[i]  = (_Float16)wdown[k * 32 + n];   return; }
    i -= s2;
    if (i < s3) { int n = i / 64,  k = i % 64;  winner_t[i] = (_Float16)winner[k * 1024 + n]; return; }
    i -= s3;
    if (i < s4) { int n = i / 1024, k = i % 1024; wouter_t[i] = (_Float16)wouter[k * 256 + n]; return; }
}

// ---------------------------------------------------------------------------
// Kernel 2: LayerNorm(x) over D=256 -> pre_f16; also x -> x_f16.
// One 256-thread block per row.
// ---------------------------------------------------------------------------
__global__ void rsb_ln_x(const float* __restrict__ x,
                         const float* __restrict__ g,
                         const float* __restrict__ bta,
                         _Float16* __restrict__ pre,
                         _Float16* __restrict__ xh)
{
    __shared__ float red0[8];
    __shared__ float red1[8];
    const int row = blockIdx.x;
    const int t = threadIdx.x;                  // 0..255
    const size_t o = (size_t)row * RD + t;
    float v = x[o];

    float s = v;
    #pragma unroll
    for (int off = 16; off >= 1; off >>= 1) s += __shfl_xor(s, off, 32);
    if ((t & 31) == 0) red0[t >> 5] = s;
    __syncthreads();
    float mean = 0.f;
    #pragma unroll
    for (int w = 0; w < 8; ++w) mean += red0[w];
    mean *= (1.0f / RD);

    float d = v - mean;
    float q = d * d;
    #pragma unroll
    for (int off = 16; off >= 1; off >>= 1) q += __shfl_xor(q, off, 32);
    if ((t & 31) == 0) red1[t >> 5] = q;
    __syncthreads();
    float var = 0.f;
    #pragma unroll
    for (int w = 0; w < 8; ++w) var += red1[w];
    var *= (1.0f / RD);

    float r = rsqrtf(var + 1e-5f);
    pre[o] = (_Float16)(d * r * g[t] + bta[t]);
    xh[o]  = (_Float16)v;
}

// ---------------------------------------------------------------------------
// Kernel 3: register-blocked WMMA GEMM.
//   C[M,Nc] = A[M,K] @ Bt[Nc,K]^T   (f16 in, f32 acc)
// One wave computes a 16 x (16*NB) C strip: the A tile is loaded once per
// K-step and multiplied against NB B tiles (A traffic amortized NB-fold;
// ~2.5 b128 loads per wmma at NB=4 instead of 4).
// Epilogue: val = (acc + bias_scale*bias[n]) * out_scale; optional exact gelu;
// mode 0: f32 store, 1: f16 store, 2: f32 store with f32 residual add.
// A-tile per lane: row m0+(l&15); halves k0+0..15 (l<16) / k0+16..31.
// B-tile per lane: row n0+(l&15) of Bt; halves k0+0..15 (l<16) / k0+16..31.
// C-tile: VGPR r, lane l -> M = r + 8*(l>=16), N = l&15.
// ---------------------------------------------------------------------------
template <int NB>
__global__ void rsb_gemm_wmma(const _Float16* __restrict__ A,
                              const _Float16* __restrict__ Bt,
                              int M, int K, int Nc,
                              const float* __restrict__ bias, float bias_scale,
                              float out_scale, int do_gelu, int mode,
                              float* __restrict__ Cf32,
                              _Float16* __restrict__ Cf16,
                              const float* __restrict__ resid)
{
    const int lane = threadIdx.x & 31;
    const int wave = threadIdx.x >> 5;
    const int ntn = Nc / (16 * NB);               // N strip blocks
    const int tile = blockIdx.x * (blockDim.x >> 5) + wave;
    if (tile >= (M >> 4) * ntn) return;           // wave-uniform: EXEC stays all-1s
    const int mt = tile / ntn, nt = tile % ntn;
    const int m0 = mt << 4, n0 = nt * (16 * NB);

    const int  rowA  = m0 + (lane & 15);
    const int  baseA = (lane < 16) ? 0 : 16;
    const int  rowB  = n0 + (lane & 15);
    const int  baseB = (lane < 16) ? 0 : 16;
    const _Float16* pa0 = A  + (size_t)rowA * K + baseA;
    const _Float16* pb0 = Bt + (size_t)rowB * K + baseB;

    v8f c[NB];
    #pragma unroll
    for (int u = 0; u < NB; ++u) c[u] = (v8f){};

    for (int k0 = 0; k0 < K; k0 += 32) {
        v8h alo = *(const v8h*)(pa0 + k0);
        v8h ahi = *(const v8h*)(pa0 + k0 + 8);
        v16h a;
        #pragma unroll
        for (int i = 0; i < 8; ++i) { a[i] = alo[i]; a[i + 8] = ahi[i]; }
        #pragma unroll
        for (int u = 0; u < NB; ++u) {
            v16h b = *(const v16h*)(pb0 + (size_t)u * 16 * K + k0);
            c[u] = __builtin_amdgcn_wmma_f32_16x16x32_f16(false, a, false, b,
                                                          (short)0, c[u], false, false);
        }
    }

    const int mBase = m0 + ((lane < 16) ? 0 : 8);
    #pragma unroll
    for (int u = 0; u < NB; ++u) {
        const int n = n0 + u * 16 + (lane & 15);
        const float bv = bias ? bias[n] * bias_scale : 0.0f;
        #pragma unroll
        for (int r = 0; r < 8; ++r) {
            float v = (c[u][r] + bv) * out_scale;
            if (do_gelu) v = 0.5f * v * (1.0f + erff(v * 0.70710678118654752f));
            const size_t o = (size_t)(mBase + r) * Nc + n;
            if (mode == 0)      Cf32[o] = v;
            else if (mode == 1) Cf16[o] = (_Float16)v;
            else                Cf32[o] = v + resid[o];
        }
    }
}

// ---------------------------------------------------------------------------
// Kernel 4: repack kv GEMM output [8192,512] into head-major
// k_f16[(b*H+h)*N + n][32] and v_f32[(b*H+h)*N + n][32].
// ---------------------------------------------------------------------------
__global__ void rsb_pack_kv(const float* __restrict__ kv,
                            _Float16* __restrict__ kf,
                            float* __restrict__ vf)
{
    size_t i = (size_t)blockIdx.x * blockDim.x + threadIdx.x;   // 2M elems
    if (i >= (size_t)RB * RH * RN * RDV) return;
    const int d = (int)(i & 31);
    size_t r = i >> 5;
    const int n = (int)(r & (RN - 1));
    r >>= 10;
    const int h = (int)(r & (RH - 1));
    const int b = (int)(r >> 3);
    const size_t src = ((size_t)(b * RN + n) * (2 * RD)) + h * RDV + d;
    kf[i] = (_Float16)kv[src];
    vf[i] = kv[src + RD];
}

// ---------------------------------------------------------------------------
// Kernel 5: scores = Q.K^T via WMMA, running argmax over j per query row.
// One wave per (b, h, 16-row i-tile). Q tile loaded once; 64 K-tiles streamed.
// argmax over the tile's 16 columns = 16-lane half-wave shfl_xor reduction;
// ties resolve to the smallest j (matches jnp.argmax).
// ---------------------------------------------------------------------------
__global__ void rsb_attn_argmax(const _Float16* __restrict__ qf,   // [B*N,32]
                                const _Float16* __restrict__ kf,   // [(b*H+h)*N+n,32]
                                int* __restrict__ idxb)            // [B*H*N]
{
    const int lane = threadIdx.x & 31;
    const int wave = threadIdx.x >> 5;
    int flat = blockIdx.x * (blockDim.x >> 5) + wave;              // 0..4095
    if (flat >= RB * RH * (RN / 16)) return;
    const int it = flat & 63; flat >>= 6;
    const int h  = flat & 7;  flat >>= 3;
    const int b  = flat;

    const int  i     = (it << 4) + (lane & 15);
    const int  baseA = (lane < 16) ? 0 : 16;
    const _Float16* pa = qf + (size_t)(b * RN + i) * RDV + baseA;
    v8h alo = *(const v8h*)pa;
    v8h ahi = *(const v8h*)(pa + 8);
    v16h a;
    #pragma unroll
    for (int t = 0; t < 8; ++t) { a[t] = alo[t]; a[t + 8] = ahi[t]; }

    const int baseB = (lane < 16) ? 0 : 16;
    const _Float16* kbase = kf + (size_t)(b * RH + h) * RN * RDV;

    float best[8];
    int   bidx[8];
    #pragma unroll
    for (int r = 0; r < 8; ++r) { best[r] = -3.0e38f; bidx[r] = 0; }

    for (int jt = 0; jt < RN / 16; ++jt) {
        const int j = (jt << 4) + (lane & 15);
        v16h bm = *(const v16h*)(kbase + (size_t)j * RDV + baseB);
        v8f c = {};
        c = __builtin_amdgcn_wmma_f32_16x16x32_f16(false, a, false, bm,
                                                   (short)0, c, false, false);
        #pragma unroll
        for (int r = 0; r < 8; ++r) {
            if (c[r] > best[r]) { best[r] = c[r]; bidx[r] = j; }
        }
    }

    #pragma unroll
    for (int r = 0; r < 8; ++r) {
        float v = best[r]; int ix = bidx[r];
        #pragma unroll
        for (int off = 8; off >= 1; off >>= 1) {
            float ov = __shfl_xor(v, off, 32);
            int  oix = __shfl_xor(ix, off, 32);
            if (ov > v || (ov == v && oix < ix)) { v = ov; ix = oix; }
        }
        bidx[r] = ix;
    }

    if ((lane & 15) == 0) {
        const int mOff = (lane < 16) ? 0 : 8;
        #pragma unroll
        for (int r = 0; r < 8; ++r)
            idxb[(size_t)(b * RH + h) * RN + (it << 4) + mOff + r] = bidx[r];
    }
}

// ---------------------------------------------------------------------------
// Kernel 6: gather hard rows of V, per-head LayerNorm over DV=32, sum over
// heads; concat with precomputed 8*(x@w_down + b_down) -> pairs_f16 [8192,64].
// One wave per row; lane == channel d.
// ---------------------------------------------------------------------------
__global__ void rsb_gather_ln_pairs(const int* __restrict__ idxb,
                                    const float* __restrict__ vf,
                                    const float* __restrict__ g,
                                    const float* __restrict__ bb,
                                    const float* __restrict__ res,   // [8192,32], already *8
                                    _Float16* __restrict__ pairs)    // [8192,64]
{
    const int lane = threadIdx.x & 31;
    const int wave = threadIdx.x >> 5;
    const int row = blockIdx.x * (blockDim.x >> 5) + wave;           // b*N+n
    if (row >= ROWS) return;
    const int b = row >> 10, n = row & (RN - 1);

    float s = 0.0f;
    #pragma unroll
    for (int h = 0; h < RH; ++h) {
        const int j = idxb[(size_t)(b * RH + h) * RN + n];
        const float v = vf[((size_t)(b * RH + h) * RN + j) * RDV + lane];
        float m = v;
        #pragma unroll
        for (int off = 16; off >= 1; off >>= 1) m += __shfl_xor(m, off, 32);
        m *= (1.0f / RDV);
        const float d = v - m;
        float q = d * d;
        #pragma unroll
        for (int off = 16; off >= 1; off >>= 1) q += __shfl_xor(q, off, 32);
        q *= (1.0f / RDV);
        s += d * rsqrtf(q + 1e-5f) * g[lane] + bb[lane];
    }
    pairs[(size_t)row * 64 + lane]      = (_Float16)s;
    pairs[(size_t)row * 64 + 32 + lane] = (_Float16)res[(size_t)row * RDV + lane];
}

// ---------------------------------------------------------------------------
// Host launcher
// ---------------------------------------------------------------------------
extern "C" void kernel_launch(void* const* d_in, const int* in_sizes, int n_in,
                              void* d_out, int out_size, void* d_ws, size_t ws_size,
                              hipStream_t stream) {
    const float* x        = (const float*)d_in[0];
    const float* w_q      = (const float*)d_in[1];
    const float* w_kv     = (const float*)d_in[2];
    const float* ln_out_g = (const float*)d_in[3];
    const float* ln_out_b = (const float*)d_in[4];
    const float* ln_rel_g = (const float*)d_in[5];
    const float* ln_rel_b = (const float*)d_in[6];
    // w_down = d_in[7] (f32 [256,32])
    const float* b_down   = (const float*)d_in[8];
    const float* w_inner  = (const float*)d_in[9];
    const float* b_inner  = (const float*)d_in[10];
    const float* w_outer  = (const float*)d_in[11];
    const float* b_outer  = (const float*)d_in[12];
    float* out = (float*)d_out;

    // bump allocator over workspace (256B aligned)
    char* wsp = (char*)d_ws;
    auto alloc = [&](size_t bytes) -> void* {
        void* p = (void*)wsp;
        wsp += (bytes + 255) & ~(size_t)255;
        return p;
    };
    _Float16* pre_f16  = (_Float16*)alloc((size_t)ROWS * RD * 2);
    _Float16* x_f16    = (_Float16*)alloc((size_t)ROWS * RD * 2);
    _Float16* wq_t     = (_Float16*)alloc((size_t)RDV * RD * 2);
    _Float16* wkv_t    = (_Float16*)alloc((size_t)(2 * RD) * RD * 2);
    _Float16* wdown_t  = (_Float16*)alloc((size_t)RDV * RD * 2);
    _Float16* winner_t = (_Float16*)alloc((size_t)RHID * (2 * RDV) * 2);
    _Float16* wouter_t = (_Float16*)alloc((size_t)RD * RHID * 2);
    _Float16* q_f16    = (_Float16*)alloc((size_t)ROWS * RDV * 2);
    float*    kv_f32   = (float*)   alloc((size_t)ROWS * (2 * RD) * 4);
    _Float16* k_f16    = (_Float16*)alloc((size_t)RB * RH * RN * RDV * 2);
    float*    v_f32    = (float*)   alloc((size_t)RB * RH * RN * RDV * 4);
    int*      idxb     = (int*)     alloc((size_t)RB * RH * RN * 4);
    float*    res_f32  = (float*)   alloc((size_t)ROWS * RDV * 4);
    _Float16* pairs    = (_Float16*)alloc((size_t)ROWS * 64 * 2);
    _Float16* mid_f16  = (_Float16*)alloc((size_t)ROWS * RHID * 2);
    (void)ws_size; (void)n_in; (void)in_sizes; (void)out_size;

    // 1) weights -> f16 transposed
    {
        const int total = 32 * 256 + 512 * 256 + 32 * 256 + 1024 * 64 + 256 * 1024;
        rsb_prep_weights<<<(total + 255) / 256, 256, 0, stream>>>(
            w_q, w_kv, (const float*)d_in[7], w_inner, w_outer,
            wq_t, wkv_t, wdown_t, winner_t, wouter_t);
    }
    // 2) LN(x) -> pre_f16, x -> x_f16
    rsb_ln_x<<<ROWS, RD, 0, stream>>>(x, ln_out_g, ln_out_b, pre_f16, x_f16);

    const int WPB = 4, TPB = 32 * WPB;
    auto gemm_blocks = [](int M, int Nc, int NB) {
        return ((M / 16) * (Nc / (16 * NB)) + 3) / 4;
    };

    // 3) q = pre @ w_q, scaled by DV^-0.5, stored f16 (Nc=32 -> NB=2)
    rsb_gemm_wmma<2><<<gemm_blocks(ROWS, RDV, 2), TPB, 0, stream>>>(
        pre_f16, wq_t, ROWS, RD, RDV,
        nullptr, 0.0f, 0.17677669529663687f, 0, 1, nullptr, q_f16, nullptr);
    // 4) kv = pre @ w_kv, f32 (Nc=512 -> NB=4)
    rsb_gemm_wmma<4><<<gemm_blocks(ROWS, 2 * RD, 4), TPB, 0, stream>>>(
        pre_f16, wkv_t, ROWS, RD, 2 * RD,
        nullptr, 0.0f, 1.0f, 0, 0, kv_f32, nullptr, nullptr);
    // 5) res = 8*(x @ w_down + b_down), f32 (Nc=32 -> NB=2)
    rsb_gemm_wmma<2><<<gemm_blocks(ROWS, RDV, 2), TPB, 0, stream>>>(
        x_f16, wdown_t, ROWS, RD, RDV,
        b_down, 1.0f, 8.0f, 0, 0, res_f32, nullptr, nullptr);
    // 6) repack k (f16, head-major) and v (f32, head-major)
    {
        const size_t total = (size_t)RB * RH * RN * RDV;
        rsb_pack_kv<<<(unsigned)((total + 255) / 256), 256, 0, stream>>>(kv_f32, k_f16, v_f32);
    }
    // 7) WMMA score tiles + argmax -> idx
    rsb_attn_argmax<<<(RB * RH * (RN / 16) + WPB - 1) / WPB, TPB, 0, stream>>>(
        q_f16, k_f16, idxb);
    // 8) gather hard V, per-head LN, head-sum, concat residual -> pairs
    rsb_gather_ln_pairs<<<(ROWS + WPB - 1) / WPB, TPB, 0, stream>>>(
        idxb, v_f32, ln_rel_g, ln_rel_b, res_f32, pairs);
    // 9) mid = gelu(pairs @ w_inner + 8*b_inner), f16 (Nc=1024 -> NB=4)
    rsb_gemm_wmma<4><<<gemm_blocks(ROWS, RHID, 4), TPB, 0, stream>>>(
        pairs, winner_t, ROWS, 2 * RDV, RHID,
        b_inner, 8.0f, 1.0f, 1, 1, nullptr, mid_f16, nullptr);
    // 10) out = mid @ w_outer + b_outer + x, f32 -> d_out (Nc=256 -> NB=4)
    rsb_gemm_wmma<4><<<gemm_blocks(ROWS, RD, 4), TPB, 0, stream>>>(
        mid_f16, wouter_t, ROWS, RHID, RD,
        b_outer, 1.0f, 1.0f, 0, 2, out, nullptr, x);
}
